// MultiHeadAttention_76897094467838
// MI455X (gfx1250) — matrix operand order repaired and checked by reading, not compiled, not run on hardware
//
#include <hip/hip_runtime.h>
#include <hip/hip_bf16.h>
#include <stdint.h>

// ---------------- problem constants ----------------
constexpr int kB  = 4;
constexpr int kT  = 2048;
constexpr int kD  = 1024;
constexpr int kH  = 16;
constexpr int kHS = 64;

typedef __attribute__((ext_vector_type(16))) __bf16 v16bf;
typedef __attribute__((ext_vector_type(8)))  float  v8f;
typedef __attribute__((ext_vector_type(4)))  unsigned int v4u;
typedef __attribute__((ext_vector_type(8)))  int    v8i_;
typedef __attribute__((ext_vector_type(4)))  int    v4i_;

#define WMMA_BF16(a, b, c) \
  __builtin_amdgcn_wmma_f32_16x16x32_bf16(false, (a), false, (b), (short)0, (c), false, false)

__device__ __forceinline__ uint16_t f32_to_bf16(float f) {
  union { float f; uint32_t u; } x; x.f = f;
  uint32_t u = x.u;
  uint32_t r = (u + 0x7FFFu + ((u >> 16) & 1u)) >> 16;  // RNE
  return (uint16_t)r;
}

union ABFrag { v16bf v; uint32_t u[8]; };

// A fragment: 16x32 bf16, element (m,k) at base[m*ld + k].
__device__ __forceinline__ v16bf load_frag_a(const uint16_t* base, int ld, int lane) {
  ABFrag r;
  const int m = lane & 15, half = lane >> 4;
  const uint16_t* row = base + (size_t)m * ld;
#pragma unroll
  for (int j = 0; j < 8; ++j) {
    const int k0 = (j < 4) ? (8 * half + 2 * j) : (16 + 8 * half + 2 * (j - 4));
    r.u[j] = *(const uint32_t*)(row + k0);
  }
  return r.v;
}

// B fragment: 32x16 bf16, element (k,n) at base[n*ld + k]  (N-major, K contiguous).
__device__ __forceinline__ v16bf load_frag_b(const uint16_t* base, int ld, int lane) {
  ABFrag r;
  const int n = lane & 15, half = lane >> 4;
  const uint16_t* row = base + (size_t)n * ld;
#pragma unroll
  for (int j = 0; j < 8; ++j) {
    const int k0 = (j < 4) ? (8 * half + 2 * j) : (16 + 8 * half + 2 * (j - 4));
    r.u[j] = *(const uint32_t*)(row + k0);
  }
  return r.v;
}

// ---- Tensor Data Mover: 2-D tile (rows x row_len bf16) global -> LDS ----
// D# per CDNA5 ISA ch.8: group0 = {count, lds_addr, global_addr[56:0], type=2},
// group1 = {wg_mask=0, data_size=1(2B), dims/strides, tile dims}.
// clang-23 toolchain: 6-arg builtin (g0, g1, g2, g3, g_ext, cpol).
__device__ __forceinline__ void tdm_load_2d(const uint16_t* gptr, uint32_t lds_off,
                                            uint32_t tile_d0, uint32_t tile_d1,
                                            uint32_t row_stride,
                                            uint32_t tensor_d0, uint32_t tensor_d1) {
  const uint64_t g = (uint64_t)(uintptr_t)gptr;
  v4u g0;
  g0[0] = 1u;                                               // count=1 (valid user D#)
  g0[1] = lds_off;                                          // lds_addr (bytes)
  g0[2] = (uint32_t)g;                                      // global_addr[31:0] -> bits 95:64
  g0[3] = ((uint32_t)(g >> 32) & 0x01FFFFFFu) | (2u << 30); // global_addr[56:32] | type=2
  v8i_ g1;
  g1[0] = (int)(1u << 16);                                  // data_size=1 (2 bytes), wg_mask=0
  g1[1] = (int)((tensor_d0 & 0xFFFFu) << 16);               // tensor_dim0[15:0]
  g1[2] = (int)(((tensor_d0 >> 16) & 0xFFFFu) |
                ((tensor_d1 & 0xFFFFu) << 16));             // tensor_dim0[31:16] | tensor_dim1[15:0]
  g1[3] = (int)(((tensor_d1 >> 16) & 0xFFFFu) |
                ((tile_d0 & 0xFFFFu) << 16));               // tensor_dim1[31:16] | tile_dim0
  g1[4] = (int)(tile_d1 & 0xFFFFu);                         // tile_dim1 ; tile_dim2=0
  g1[5] = (int)row_stride;                                  // tensor_dim0_stride[31:0]
  g1[6] = 0;
  g1[7] = 0;
  const v4i_ z4 = {0, 0, 0, 0};
  const v8i_ z8 = {0, 0, 0, 0, 0, 0, 0, 0};
  __builtin_amdgcn_tensor_load_to_lds(g0, g1, z4, z4, z8, 0);
}

// ---------------- conversion kernels ----------------
__global__ void conv_x_kernel(const float* __restrict__ x, uint16_t* __restrict__ xb, int n) {
  int i = blockIdx.x * 256 + threadIdx.x;
  if (i < n) xb[i] = f32_to_bf16(x[i]);
}

// W [H, D, HS] fp32 -> Wt [H, HS, D] bf16 (K=D contiguous per output row)
__global__ void conv_w_kernel(const float* __restrict__ w, uint16_t* __restrict__ wt, int n) {
  int i = blockIdx.x * 256 + threadIdx.x;
  if (i >= n) return;
  int e = i % kHS;
  int r = i / kHS;
  int d = r % kD;
  int h = r / kD;
  wt[((size_t)(h * kHS + e)) * kD + d] = f32_to_bf16(w[i]);
}

// Wp [K=1024, N=1024] fp32 -> Wpt [N, K] bf16
__global__ void conv_wp_kernel(const float* __restrict__ wp, uint16_t* __restrict__ wpt) {
  int i = blockIdx.x * 256 + threadIdx.x;   // i = k*kD + n
  int n = i % kD;
  int k = i / kD;
  wpt[(size_t)n * kD + k] = f32_to_bf16(wp[i]);
}

// ---------------- QKV projection GEMM ----------------
// one wave -> 64 rows x 64 cols (full HS): 16 accumulators, 16 WMMAs per K-step
__global__ __launch_bounds__(256) void qkv_kernel(
    const uint16_t* __restrict__ xb,
    const uint16_t* __restrict__ wqt, const uint16_t* __restrict__ wkt,
    const uint16_t* __restrict__ wvt,
    uint16_t* __restrict__ q, uint16_t* __restrict__ kk, uint16_t* __restrict__ vt) {
  const int lane = threadIdx.x & 31;
  const int wave = threadIdx.x >> 5;
  int gid = blockIdx.x * 8 + wave;            // ((tgt*4 + b)*16 + h)*32 + mt
  const int mt = gid & 31;  gid >>= 5;        // 64-row strip of T
  const int h  = gid & 15;  gid >>= 4;
  const int b  = gid & 3;   gid >>= 2;
  const int tgt = gid;                        // 0=Q 1=K 2=V

  const uint16_t* wsel  = (tgt == 0) ? wqt : (tgt == 1) ? wkt : wvt;
  const uint16_t* wbase = wsel + (size_t)h * kHS * kD;
  const uint16_t* abase = xb + ((size_t)(b * kT + mt * 64)) * kD;

  v8f acc[4][4] = {};
#pragma unroll 1
  for (int k0 = 0; k0 < kD; k0 += 32) {
    v16bf a[4];
#pragma unroll
    for (int r = 0; r < 4; ++r)
      a[r] = load_frag_a(abase + (size_t)(r * 16) * kD + k0, kD, lane);
#pragma unroll
    for (int nt = 0; nt < 4; ++nt) {
      v16bf bf = load_frag_b(wbase + (size_t)(nt * 16) * kD + k0, kD, lane);
#pragma unroll
      for (int r = 0; r < 4; ++r)
        acc[r][nt] = WMMA_BF16(a[r], bf, acc[r][nt]);
    }
  }

  const int n = lane & 15, half = lane >> 4;
#pragma unroll
  for (int r = 0; r < 4; ++r)
#pragma unroll
    for (int nt = 0; nt < 4; ++nt)
#pragma unroll
      for (int v = 0; v < 8; ++v) {
        const int m = v + 8 * half;
        const int t = mt * 64 + r * 16 + m;
        const int e = nt * 16 + n;
        const uint16_t h16 = f32_to_bf16(acc[r][nt][v]);
        if (tgt == 2) {
          vt[((size_t)((b * kH + h) * kHS + e)) * kT + t] = h16;   // V^T [B,H,HS,T]
        } else {
          uint16_t* dst = (tgt == 0) ? q : kk;
          dst[((size_t)((b * kH + h) * kT + t)) * kHS + e] = h16;  // [B,H,T,HS]
        }
      }
}

// ---------------- flash attention (double-buffered TDM-staged K/V tiles) ----------------
#define ATT_WPB 4
__global__ __launch_bounds__(128) void attn_kernel(
    const uint16_t* __restrict__ q, const uint16_t* __restrict__ kkey,
    const uint16_t* __restrict__ vt, uint16_t* __restrict__ ob) {
  __shared__ uint16_t kbuf[ATT_WPB][2][32 * 64];   // K tile ping-pong: [s_local][e]
  __shared__ uint16_t vbuf[ATT_WPB][2][64 * 32];   // V^T tile ping-pong: [e][s_local]
  __shared__ uint16_t pbuf[ATT_WPB][16 * 32];      // P staging

  const int lane = threadIdx.x & 31;
  const int wave = threadIdx.x >> 5;
  int gid = blockIdx.x * ATT_WPB + wave;           // ((b*16 + h)*128) + qt
  const int qt = gid & 127; gid >>= 7;
  const int h  = gid & 15;  gid >>= 4;
  const int b  = gid;

  const int q_base = qt * 16;
  const uint16_t* qh = q    + (size_t)(b * kH + h) * kT * kHS;
  const uint16_t* kh = kkey + (size_t)(b * kH + h) * kT * kHS;
  const uint16_t* vh = vt   + (size_t)(b * kH + h) * kHS * kT;

  const uint32_t k_lds[2] = {(uint32_t)(uintptr_t)&kbuf[wave][0][0],
                             (uint32_t)(uintptr_t)&kbuf[wave][1][0]};
  const uint32_t v_lds[2] = {(uint32_t)(uintptr_t)&vbuf[wave][0][0],
                             (uint32_t)(uintptr_t)&vbuf[wave][1][0]};

  const v16bf aq0 = load_frag_a(qh + (size_t)q_base * kHS + 0,  kHS, lane);
  const v16bf aq1 = load_frag_a(qh + (size_t)q_base * kHS + 32, kHS, lane);

  v8f acc[4] = {};
  float row_max[8], row_sum[8];
#pragma unroll
  for (int v = 0; v < 8; ++v) { row_max[v] = -1e30f; row_sum[v] = 0.0f; }

  const int n = lane & 15, half = lane >> 4;
  const float scale = 0.125f;                      // HS^-0.5

  const int nchunks = (q_base + 15) / 32 + 1;
  // prologue: stage chunk 0
  tdm_load_2d(kh, k_lds[0], /*tile*/64, 32, /*stride*/kHS, kHS, kT);
  tdm_load_2d(vh, v_lds[0], /*tile*/32, 64, /*stride*/kT,  kT,  kHS);

#pragma unroll 1
  for (int i = 0; i < nchunks; ++i) {
    const int s0 = i * 32;
    const int cur = i & 1;
    const bool more = (i + 1) < nchunks;
    if (more) {
      const int sn = s0 + 32;
      tdm_load_2d(kh + (size_t)sn * kHS, k_lds[cur ^ 1], 64, 32, kHS, kHS, kT);
      tdm_load_2d(vh + sn,               v_lds[cur ^ 1], 32, 64, kT,  kT,  kHS);
      __builtin_amdgcn_s_wait_tensorcnt((short)2);   // chunk i landed (TDM in-order)
    } else {
      __builtin_amdgcn_s_wait_tensorcnt((short)0);
    }
    asm volatile("" ::: "memory");

    // ---- scores: two 16x16 tiles of S = Q K^T, K-frags from LDS ----
    float sc[2][8];
#pragma unroll
    for (int kt = 0; kt < 2; ++kt) {
      v16bf b0 = load_frag_b(&kbuf[wave][cur][kt * 16 * 64] + 0,  64, lane);
      v16bf b1 = load_frag_b(&kbuf[wave][cur][kt * 16 * 64] + 32, 64, lane);
      v8f c = {};
      c = WMMA_BF16(aq0, b0, c);
      c = WMMA_BF16(aq1, b1, c);
      const int s = s0 + kt * 16 + n;
#pragma unroll
      for (int v = 0; v < 8; ++v) {
        const int t = q_base + v + 8 * half;
        float val = c[v] * scale;
        if (s > t) val = -1e30f;                   // causal mask
        sc[kt][v] = val;
      }
    }
    // ---- online softmax ----
#pragma unroll
    for (int v = 0; v < 8; ++v) {
      float mx = fmaxf(sc[0][v], sc[1][v]);
#pragma unroll
      for (int msk = 1; msk < 16; msk <<= 1) mx = fmaxf(mx, __shfl_xor(mx, msk, 32));
      const float m_new = fmaxf(row_max[v], mx);
      const float alpha = __expf(row_max[v] - m_new);
      row_max[v] = m_new;
      const float p0 = __expf(sc[0][v] - m_new);
      const float p1 = __expf(sc[1][v] - m_new);
      sc[0][v] = p0; sc[1][v] = p1;
      float rs = p0 + p1;
#pragma unroll
      for (int msk = 1; msk < 16; msk <<= 1) rs += __shfl_xor(rs, msk, 32);
      row_sum[v] = row_sum[v] * alpha + rs;
#pragma unroll
      for (int j = 0; j < 4; ++j) acc[j][v] *= alpha;
    }
    // ---- P -> LDS -> A fragment ----
#pragma unroll
    for (int kt = 0; kt < 2; ++kt)
#pragma unroll
      for (int v = 0; v < 8; ++v) {
        const int m = v + 8 * half;
        pbuf[wave][m * 32 + kt * 16 + n] = f32_to_bf16(sc[kt][v]);
      }
    asm volatile("s_wait_dscnt 0" ::: "memory");
    v16bf ap = load_frag_a(&pbuf[wave][0], 32, lane);
    // ---- O += P V, V-frags from LDS ----
#pragma unroll
    for (int j = 0; j < 4; ++j) {
      v16bf bv = load_frag_b(&vbuf[wave][cur][(j * 16) * 32], 32, lane);
      acc[j] = WMMA_BF16(ap, bv, acc[j]);
    }
  }

  // normalize, store concat-head bf16 O [B, T, H*HS]
#pragma unroll
  for (int j = 0; j < 4; ++j)
#pragma unroll
    for (int v = 0; v < 8; ++v) {
      const int m = v + 8 * half;
      const int t = q_base + m;
      const float o = acc[j][v] / row_sum[v];
      ob[(size_t)(b * kT + t) * kD + h * 64 + j * 16 + n] = f32_to_bf16(o);
    }
}

// ---------------- output projection GEMM + bias ----------------
// one wave -> 64 rows x 64 cols
__global__ __launch_bounds__(256) void proj_kernel(
    const uint16_t* __restrict__ ob, const uint16_t* __restrict__ wpt,
    const float* __restrict__ bp, float* __restrict__ out) {
  const int lane = threadIdx.x & 31;
  const int wave = threadIdx.x >> 5;
  int gid = blockIdx.x * 8 + wave;            // mt*16 + ntg
  const int ntg = gid & 15; gid >>= 4;        // 64-column group
  const int mt  = gid;                        // 0..127 (64-row strip of B*T rows)

  const uint16_t* abase = ob + (size_t)(mt * 64) * kD;
  v8f acc[4][4] = {};
#pragma unroll 1
  for (int k0 = 0; k0 < kD; k0 += 32) {
    v16bf a[4];
#pragma unroll
    for (int r = 0; r < 4; ++r)
      a[r] = load_frag_a(abase + (size_t)(r * 16) * kD + k0, kD, lane);
#pragma unroll
    for (int j = 0; j < 4; ++j) {
      v16bf bf = load_frag_b(wpt + (size_t)(ntg * 64 + j * 16) * kD + k0, kD, lane);
#pragma unroll
      for (int r = 0; r < 4; ++r)
        acc[r][j] = WMMA_BF16(a[r], bf, acc[r][j]);
    }
  }

  const int n = lane & 15, half = lane >> 4;
#pragma unroll
  for (int j = 0; j < 4; ++j) {
    const int col = ntg * 64 + j * 16 + n;
    const float bias = bp[col];
#pragma unroll
    for (int r = 0; r < 4; ++r)
#pragma unroll
      for (int v = 0; v < 8; ++v) {
        const int row = mt * 64 + r * 16 + v + 8 * half;
        out[(size_t)row * kD + col] = acc[r][j][v] + bias;
      }
  }
}

// ---------------- launcher ----------------
extern "C" void kernel_launch(void* const* d_in, const int* in_sizes, int n_in,
                              void* d_out, int out_size, void* d_ws, size_t ws_size,
                              hipStream_t stream) {
  (void)in_sizes; (void)n_in; (void)out_size; (void)ws_size;
  const float* x  = (const float*)d_in[0];
  const float* Wq = (const float*)d_in[1];
  const float* Wk = (const float*)d_in[2];
  const float* Wv = (const float*)d_in[3];
  const float* Wp = (const float*)d_in[4];
  const float* bp = (const float*)d_in[5];
  float* out = (float*)d_out;

  const size_t XB = (size_t)kB * kT * kD * 2;        // 16 MB
  const size_t WT = (size_t)kH * kHS * kD * 2;       // 2 MB
  const size_t QS = (size_t)kB * kH * kT * kHS * 2;  // 16 MB
  char* w = (char*)d_ws;
  uint16_t* xb  = (uint16_t*)(w);
  uint16_t* wqt = (uint16_t*)(w + XB);
  uint16_t* wkt = (uint16_t*)(w + XB + WT);
  uint16_t* wvt = (uint16_t*)(w + XB + 2 * WT);
  uint16_t* wpt = (uint16_t*)(w + XB + 3 * WT);
  uint16_t* qb  = (uint16_t*)(w + XB + 4 * WT);
  uint16_t* kb  = (uint16_t*)(w + XB + 4 * WT + QS);
  uint16_t* vtb = (uint16_t*)(w + XB + 4 * WT + 2 * QS);
  uint16_t* obf = (uint16_t*)(w + XB + 4 * WT + 3 * QS);

  {
    const int nx = kB * kT * kD;
    conv_x_kernel<<<(nx + 255) / 256, 256, 0, stream>>>(x, xb, nx);
    const int nw = kH * kD * kHS;
    conv_w_kernel<<<(nw + 255) / 256, 256, 0, stream>>>(Wq, wqt, nw);
    conv_w_kernel<<<(nw + 255) / 256, 256, 0, stream>>>(Wk, wkt, nw);
    conv_w_kernel<<<(nw + 255) / 256, 256, 0, stream>>>(Wv, wvt, nw);
    conv_wp_kernel<<<(kD * kD + 255) / 256, 256, 0, stream>>>(Wp, wpt);
  }
  // QKV: 3*B*H*(T/64) = 6144 waves, 8/block
  qkv_kernel<<<(3 * kB * kH * (kT / 64)) / 8, 256, 0, stream>>>(xb, wqt, wkt, wvt, qb, kb, vtb);
  // attention: B*H*(T/16) = 8192 waves, 4/block
  attn_kernel<<<(kB * kH * (kT / 16)) / ATT_WPB, 32 * ATT_WPB, 0, stream>>>(qb, kb, vtb, obf);
  // out-proj: (B*T/64)*(D/64) = 2048 waves, 8/block
  proj_kernel<<<((kB * kT / 64) * (kD / 64)) / 8, 256, 0, stream>>>(obf, wpt, bp, out);
}
